// FCOS_14851996909952
// MI455X (gfx1250) — matrix operand order repaired and checked by reading, not compiled, not run on hardware
//
#include <hip/hip_runtime.h>

// ---------------------------------------------------------------------------
// FCOS/YOLO-FPN head for MI455X (gfx1250, wave32).
// Convs = implicit GEMM on v_wmma_f32_16x16x32_bf16, LDS-staged 64x64x64 tiles.
// ---------------------------------------------------------------------------

typedef __attribute__((ext_vector_type(16))) __bf16 v16bf;
typedef __attribute__((ext_vector_type(8)))  __bf16 v8bf;
typedef __attribute__((ext_vector_type(8)))  float  v8f;

#define NTOT 21760   // 128^2 + 64^2 + 32^2 + 16^2

__device__ __forceinline__ __bf16 f2bf(float f) {
  union { float f; unsigned u; } a; a.f = f;
  unsigned r = a.u + 0x7FFFu + ((a.u >> 16) & 1u);   // round-to-nearest-even
  unsigned short h = (unsigned short)(r >> 16);
  return __builtin_bit_cast(__bf16, h);
}

__device__ __forceinline__ v8f wmma_bf16(v16bf a, v16bf b, v8f c) {
  return __builtin_amdgcn_wmma_f32_16x16x32_bf16(
      false, a, false, b, (short)0, c, false, false);
}

// ---------------------------------------------------------------------------
// Implicit-GEMM conv:  Y[co][p] = leaky( sum_k W[co][k] * im2col(X)[k][p] + b )
// Block = 128 threads (4 waves). Block tile: 64 out-channels x 64 pixels.
// Wave (wr,wc) owns a 32x32 sub-tile = 2x2 WMMA accumulators.
// K-loop in steps of 64 (two 16x16x32 bf16 WMMA sub-chunks) -> 8 wmma/iter.
// ---------------------------------------------------------------------------
__global__ __launch_bounds__(128) void conv_wmma(
    const float* __restrict__ X, const float* __restrict__ Wt,
    const float* __restrict__ Bs, float* __restrict__ Y,
    int Cin, int Hin, int Win, int Cout, int Hout, int Wout,
    int kk, int stride, int pad, int leaky)
{
  const int P   = Hout * Wout;
  const int KK  = kk * kk;
  const int K   = Cin * KK;              // multiple of 64 for every layer here
  const int pBase  = blockIdx.x * 64;
  const int coBase = blockIdx.y * 64;

  const int t    = threadIdx.x;
  const int lane = t & 31;
  const int wave = t >> 5;
  const int lo   = lane & 15;
  const int hi   = lane >> 4;
  const int wr   = wave >> 1;            // M sub-tile pair (0..1)
  const int wc   = wave & 1;             // N sub-tile pair (0..1)

  // rows padded to 72 bf16 (144 B = 36 dwords): conflict-free, 16B-aligned
  __shared__ __align__(16) __bf16 Als[64][72];   // [m][k]  (weights)
  __shared__ __align__(16) __bf16 Bls[64][72];   // [n][k]  (im2col, transposed)

  v8f acc00 = {}, acc01 = {}, acc10 = {}, acc11 = {};

  // cooperative-stage assignment: thread t owns row (t>>1), 32-wide k-chunk
  const int sr = t >> 1;                 // 0..63
  const int sk = (t & 1) * 32;           // 0 | 32
  const int pN = pBase + sr;
  const int oy = pN / Wout, ox = pN % Wout;
  const int coA = coBase + sr;
  const float* wrow = Wt + (size_t)coA * K;

  for (int k0 = 0; k0 < K; k0 += 64) {
    // ---- stage A: weights -> bf16 LDS (float4 vector loads) -----------
    {
      const float* wp = wrow + k0 + sk;
      #pragma unroll
      for (int e = 0; e < 32; e += 4) {
        float4 v4 = (coA < Cout) ? *(const float4*)(wp + e)
                                 : make_float4(0.f, 0.f, 0.f, 0.f);
        Als[sr][sk + e + 0] = f2bf(v4.x);
        Als[sr][sk + e + 1] = f2bf(v4.y);
        Als[sr][sk + e + 2] = f2bf(v4.z);
        Als[sr][sk + e + 3] = f2bf(v4.w);
      }
      if (coA < Cout && (k0 + 64) < K)
        __builtin_prefetch(wp + 64, 0, 0);        // global_prefetch_b8
    }
    // ---- stage B: im2col gather -> bf16 LDS (transposed [n][k]) -------
    {
      for (int e = 0; e < 32; ++e) {
        int rk  = k0 + sk + e;
        int ci  = rk / KK;
        int r2  = rk - ci * KK;
        int kh  = r2 / kk;
        int kw2 = r2 - kh * kk;
        int iy  = oy * stride + kh  - pad;
        int ix  = ox * stride + kw2 - pad;
        float v = 0.0f;
        if ((unsigned)iy < (unsigned)Hin && (unsigned)ix < (unsigned)Win)
          v = X[(size_t)ci * Hin * Win + iy * Win + ix];
        Bls[sr][sk + e] = f2bf(v);
      }
    }
    __syncthreads();

    // ---- 2x2 tiles x 2 k-sub-chunks = 8 WMMA --------------------------
    // A frag (16x32): M=lane%16; halves 0..7 -> K=kb+8*hi.., 8..15 -> K=kb+16+8*hi..
    // B frag (32x16): N=lane%16; halves 0..7 -> K=kb+8*hi.., 8..15 -> K=kb+16+8*hi..
    union U { v16bf v; v8bf h[2]; };
    const int am0 = 32 * wr + lo, am1 = am0 + 16;
    const int bn0 = 32 * wc + lo, bn1 = bn0 + 16;
    #pragma unroll
    for (int s = 0; s < 2; ++s) {
      const int kb = 32 * s;
      U a0, a1, b0, b1;
      a0.h[0] = *(const v8bf*)&Als[am0][kb + 8 * hi];
      a0.h[1] = *(const v8bf*)&Als[am0][kb + 16 + 8 * hi];
      a1.h[0] = *(const v8bf*)&Als[am1][kb + 8 * hi];
      a1.h[1] = *(const v8bf*)&Als[am1][kb + 16 + 8 * hi];
      b0.h[0] = *(const v8bf*)&Bls[bn0][kb + 8 * hi];
      b0.h[1] = *(const v8bf*)&Bls[bn0][kb + 16 + 8 * hi];
      b1.h[0] = *(const v8bf*)&Bls[bn1][kb + 8 * hi];
      b1.h[1] = *(const v8bf*)&Bls[bn1][kb + 16 + 8 * hi];
      acc00 = wmma_bf16(a0.v, b0.v, acc00);
      acc01 = wmma_bf16(a0.v, b1.v, acc01);
      acc10 = wmma_bf16(a1.v, b0.v, acc10);
      acc11 = wmma_bf16(a1.v, b1.v, acc11);
    }
    __syncthreads();
  }

  // ---- epilogue: bias + leaky ReLU, masked store ---------------------------
  // C/D layout: VGPR rr -> M = rr + 8*hi, N = lane%16
  #pragma unroll
  for (int i = 0; i < 2; ++i) {
    #pragma unroll
    for (int j = 0; j < 2; ++j) {
      const v8f* ap = (i == 0) ? (j == 0 ? &acc00 : &acc01)
                               : (j == 0 ? &acc10 : &acc11);
      const int p = pBase + 32 * wc + 16 * j + lo;
      #pragma unroll
      for (int rr = 0; rr < 8; ++rr) {
        int co = coBase + 32 * wr + 16 * i + 8 * hi + rr;
        if (co < Cout) {
          float y = (*ap)[rr] + Bs[co];
          if (leaky) y = (y > 0.0f) ? y : 0.1f * y;
          Y[(size_t)co * P + p] = y;
        }
      }
    }
  }
}

// ---------------------------------------------------------------------------
__global__ void copyK(const float* __restrict__ src, float* __restrict__ dst, int n) {
  int i = blockIdx.x * blockDim.x + threadIdx.x;
  if (i < n) dst[i] = src[i];
}

// align-corners bilinear (matches jnp.linspace(0, h-1, H) sampling)
__global__ void upsampleK(const float* __restrict__ src, float* __restrict__ dst,
                          int C, int h, int w, int H, int W) {
  int i = blockIdx.x * blockDim.x + threadIdx.x;
  int total = C * H * W;
  if (i >= total) return;
  int x = i % W; int y = (i / W) % H; int c = i / (W * H);
  float ys = (H > 1) ? (float)y * (float)(h - 1) / (float)(H - 1) : 0.f;
  float xs = (W > 1) ? (float)x * (float)(w - 1) / (float)(W - 1) : 0.f;
  int y0 = (int)floorf(ys); y0 = y0 < 0 ? 0 : (y0 > h - 2 ? h - 2 : y0);
  int x0 = (int)floorf(xs); x0 = x0 < 0 ? 0 : (x0 > w - 2 ? w - 2 : x0);
  float wy = ys - (float)y0, wx = xs - (float)x0;
  int y1 = y0 + 1 < h - 1 ? y0 + 1 : h - 1;
  int x1 = x0 + 1 < w - 1 ? x0 + 1 : w - 1;
  const float* s = src + (size_t)c * h * w;
  float v00 = s[y0 * w + x0], v01 = s[y0 * w + x1];
  float v10 = s[y1 * w + x0], v11 = s[y1 * w + x1];
  float top = v00 * (1.f - wx) + v01 * wx;
  float bot = v10 * (1.f - wx) + v11 * wx;
  dst[i] = top * (1.f - wy) + bot * wy;
}

// ---------------------------------------------------------------------------
// Decode: sigmoid class scores, argmax, exp-box regression + grid, clip.
// pred buffer = [p3(26x16384) | p4(26x4096) | p5(26x1024) | p6(26x256)],
// each scale channel-major.
// ---------------------------------------------------------------------------
__global__ void decodeK(const float* __restrict__ pred,
                        float* __restrict__ out,
                        float* __restrict__ nmsb,
                        float* __restrict__ validf) {
  int t = blockIdx.x * blockDim.x + threadIdx.x;
  if (t >= NTOT) return;
  int s, local, n, base;
  if      (t < 16384) { s = 8;  local = t;         n = 128; base = 0; }
  else if (t < 20480) { s = 16; local = t - 16384; n = 64;  base = 26 * 16384; }
  else if (t < 21504) { s = 32; local = t - 20480; n = 32;  base = 26 * (16384 + 4096); }
  else                { s = 64; local = t - 21504; n = 16;  base = 26 * (16384 + 4096 + 1024); }
  int Ps = n * n;
  const float* tp = pred + base;

  float best = -1.0f; int bi = 0;
  for (int c = 0; c < 20; ++c) {
    float v  = tp[(size_t)(1 + c) * Ps + local];
    float sc = 1.0f / (1.0f + __expf(-v));
    if (sc > best) { best = sc; bi = c; }
  }
  int gy = local / n, gx = local % n;
  float cx = (float)gx * (float)s + 0.5f * (float)s;
  float cy = (float)gy * (float)s + 0.5f * (float)s;
  float l0 = __expf(tp[(size_t)22 * Ps + local]);
  float l1 = __expf(tp[(size_t)23 * Ps + local]);
  float l2 = __expf(tp[(size_t)24 * Ps + local]);
  float l3 = __expf(tp[(size_t)25 * Ps + local]);
  float bx1 = cx - l0, by1 = cy - l1, bx2 = cx + l2, by2 = cy + l3;

  float off = (float)bi * 2048.0f;                  // cls * 2*SIZE
  nmsb[4 * t + 0] = bx1 + off;
  nmsb[4 * t + 1] = by1 + off;
  nmsb[4 * t + 2] = bx2 + off;
  nmsb[4 * t + 3] = by2 + off;

  float valid = (best >= 0.01f) ? 1.0f : 0.0f;
  validf[t] = valid;

  out[4 * t + 0] = fminf(fmaxf(bx1, 0.f), 1023.f) * (1.0f / 1024.0f);
  out[4 * t + 1] = fminf(fmaxf(by1, 0.f), 1023.f) * (1.0f / 1024.0f);
  out[4 * t + 2] = fminf(fmaxf(bx2, 0.f), 1023.f) * (1.0f / 1024.0f);
  out[4 * t + 3] = fminf(fmaxf(by2, 0.f), 1023.f) * (1.0f / 1024.0f);
  out[4 * NTOT + t] = best;
  out[5 * NTOT + t] = (float)bi;
  out[6 * NTOT + t] = valid;   // provisional keep (NMS overwrites)
}

// Parallel NMS: suppress i if any valid higher-ranked j overlaps > thr.
__global__ void nmsK(const float* __restrict__ nmsb,
                     const float* __restrict__ scores,
                     const float* __restrict__ validf,
                     float* __restrict__ keep) {
  int i = blockIdx.x * blockDim.x + threadIdx.x;
  if (i >= NTOT) return;
  if (validf[i] == 0.0f) { keep[i] = 0.0f; return; }
  float x1 = nmsb[4 * i], y1 = nmsb[4 * i + 1];
  float x2 = nmsb[4 * i + 2], y2 = nmsb[4 * i + 3];
  float ai = (x2 - x1) * (y2 - y1);
  float si = scores[i];
  float k = 1.0f;
  for (int j = 0; j < NTOT; ++j) {
    if (validf[j] == 0.0f) continue;
    float sj = scores[j];
    if (sj < si || (sj == si && j >= i)) continue;   // rank order
    float jx1 = nmsb[4 * j], jy1 = nmsb[4 * j + 1];
    float jx2 = nmsb[4 * j + 2], jy2 = nmsb[4 * j + 3];
    float xx1 = fmaxf(x1, jx1), yy1 = fmaxf(y1, jy1);
    float xx2 = fminf(x2, jx2), yy2 = fminf(y2, jy2);
    float inter = fmaxf(1e-28f, xx2 - xx1) * fmaxf(1e-28f, yy2 - yy1);
    float aj = (jx2 - jx1) * (jy2 - jy1);
    float ovr = inter / (ai + aj - inter);
    if (ovr > 0.5f) { k = 0.0f; break; }
  }
  keep[i] = k;
}

// ---------------------------------------------------------------------------
// Host-side orchestration.
//
// d_in flattening (setup_inputs insertion order, leaves of nested dict):
//  0:C3  1:C4  2:C5
//  3,4   conv_3x3_6 {w,b}
//  5..10 conv_set_6 (w,b)x3      11..16 conv_set_5
// 17..22 conv_set_4              23..28 conv_set_3
// 29,30  conv_1x1_5              31,32  conv_1x1_4
// 33..36 pred_3 (w0,b0,w1,b1)    37..40 pred_4
// 41..44 pred_5                  45..48 pred_6
// ---------------------------------------------------------------------------
extern "C" void kernel_launch(void* const* d_in, const int* in_sizes, int n_in,
                              void* d_out, int out_size, void* d_ws, size_t ws_size,
                              hipStream_t stream) {
  (void)in_sizes; (void)n_in; (void)out_size; (void)ws_size;
  const float* C3 = (const float*)d_in[0];
  const float* C4 = (const float*)d_in[1];
  const float* C5 = (const float*)d_in[2];
  auto W = [&](int i) { return (const float*)d_in[i]; };

  float* ws = (float*)d_ws;
  size_t o = 0;
  auto alloc = [&](size_t nf) { float* p = ws + o; o += nf; return p; };

  // pred outputs, channel-major per scale, in pixel-concatenation order
  float* PRED  = alloc((size_t)26 * NTOT);
  float* PRED3 = PRED;
  float* PRED4 = PRED + (size_t)26 * 16384;
  float* PRED5 = PRED4 + (size_t)26 * 4096;
  float* PRED6 = PRED5 + (size_t)26 * 1024;

  float* C6  = alloc(1024 * 256);
  float* T6A = alloc(512 * 256);
  float* T6B = alloc(1024 * 256);
  float* P6  = alloc(512 * 256);
  float* H6  = alloc(1024 * 256);
  float* T5A = alloc(256 * 1024);
  float* T5B = alloc(512 * 1024);
  float* P5  = alloc(256 * 1024);
  float* P5C = alloc(128 * 1024);
  float* H5  = alloc(512 * 1024);
  float* X4  = alloc((size_t)384 * 4096);
  float* T4A = alloc(128 * 4096);
  float* T4B = alloc(256 * 4096);
  float* P4  = alloc(128 * 4096);
  float* P4C = alloc(64 * 4096);
  float* H4  = alloc(256 * 4096);
  float* X3  = alloc((size_t)192 * 16384);
  float* T3A = alloc((size_t)64 * 16384);
  float* T3B = alloc((size_t)128 * 16384);
  float* P3  = alloc((size_t)64 * 16384);
  float* H3  = alloc((size_t)128 * 16384);
  float* NMSB  = alloc((size_t)NTOT * 4);
  float* VALID = alloc(NTOT);

  auto conv = [&](const float* Xp, const float* Wp, const float* Bp, float* Yp,
                  int Cin, int Hin, int Win, int Cout, int kk, int stride, int leaky) {
    int pad  = kk / 2;
    int Hout = (Hin + 2 * pad - kk) / stride + 1;
    int Wout = (Win + 2 * pad - kk) / stride + 1;
    dim3 grid((Hout * Wout) / 64, (Cout + 63) / 64);
    conv_wmma<<<grid, 128, 0, stream>>>(Xp, Wp, Bp, Yp, Cin, Hin, Win,
                                        Cout, Hout, Wout, kk, stride, pad, leaky);
  };

  // ---- P6 branch -----------------------------------------------------------
  conv(C5,  W(3),  W(4),  C6,   512, 32, 32, 1024, 3, 2, 1);   // C6 = cbl s2
  conv(C6,  W(5),  W(6),  T6A, 1024, 16, 16,  512, 1, 1, 1);
  conv(T6A, W(7),  W(8),  T6B,  512, 16, 16, 1024, 3, 1, 1);
  conv(T6B, W(9),  W(10), P6,  1024, 16, 16,  512, 1, 1, 1);
  conv(P6,  W(45), W(46), H6,   512, 16, 16, 1024, 3, 1, 1);   // pred_6 head
  conv(H6,  W(47), W(48), PRED6,1024,16, 16,   26, 1, 1, 0);

  // ---- P5 branch -----------------------------------------------------------
  conv(C5,  W(11), W(12), T5A,  512, 32, 32,  256, 1, 1, 1);
  conv(T5A, W(13), W(14), T5B,  256, 32, 32,  512, 3, 1, 1);
  conv(T5B, W(15), W(16), P5,   512, 32, 32,  256, 1, 1, 1);
  conv(P5,  W(41), W(42), H5,   256, 32, 32,  512, 3, 1, 1);   // pred_5 head
  conv(H5,  W(43), W(44), PRED5,512, 32, 32,   26, 1, 1, 0);
  conv(P5,  W(29), W(30), P5C,  256, 32, 32,  128, 1, 1, 1);   // conv_1x1_5

  // ---- P4 branch: X4 = concat(C4, up(P5C)) --------------------------------
  copyK<<<(256 * 4096 + 255) / 256, 256, 0, stream>>>(C4, X4, 256 * 4096);
  upsampleK<<<(128 * 64 * 64 + 255) / 256, 256, 0, stream>>>(
      P5C, X4 + (size_t)256 * 4096, 128, 32, 32, 64, 64);
  conv(X4,  W(17), W(18), T4A,  384, 64, 64,  128, 1, 1, 1);
  conv(T4A, W(19), W(20), T4B,  128, 64, 64,  256, 3, 1, 1);
  conv(T4B, W(21), W(22), P4,   256, 64, 64,  128, 1, 1, 1);
  conv(P4,  W(37), W(38), H4,   128, 64, 64,  256, 3, 1, 1);   // pred_4 head
  conv(H4,  W(39), W(40), PRED4,256, 64, 64,   26, 1, 1, 0);
  conv(P4,  W(31), W(32), P4C,  128, 64, 64,   64, 1, 1, 1);   // conv_1x1_4

  // ---- P3 branch: X3 = concat(C3, up(P4C)) --------------------------------
  copyK<<<(128 * 16384 + 255) / 256, 256, 0, stream>>>(C3, X3, 128 * 16384);
  upsampleK<<<(64 * 128 * 128 + 255) / 256, 256, 0, stream>>>(
      P4C, X3 + (size_t)128 * 16384, 64, 64, 64, 128, 128);
  conv(X3,  W(23), W(24), T3A,  192, 128, 128,  64, 1, 1, 1);
  conv(T3A, W(25), W(26), T3B,   64, 128, 128, 128, 3, 1, 1);
  conv(T3B, W(27), W(28), P3,   128, 128, 128,  64, 1, 1, 1);
  conv(P3,  W(33), W(34), H3,    64, 128, 128, 128, 3, 1, 1);  // pred_3 head
  conv(H3,  W(35), W(36), PRED3,128, 128, 128,  26, 1, 1, 0);

  // ---- decode + NMS --------------------------------------------------------
  float* out = (float*)d_out;
  decodeK<<<(NTOT + 255) / 256, 256, 0, stream>>>(PRED, out, NMSB, VALID);
  nmsK<<<(NTOT + 255) / 256, 256, 0, stream>>>(NMSB, out + 4 * NTOT, VALID,
                                               out + 6 * NTOT);
}
